// AttentionRNN_79224966742708
// MI455X (gfx1250) — matrix operand order repaired and checked by reading, not compiled, not run on hardware
//
#include <hip/hip_runtime.h>
#include <hip/hip_bf16.h>

// ---------------------------------------------------------------------------
// CDNA5 (gfx1250) wave32 WMMA implementation of the attention-RNN LM head.
// All GEMMs: bf16 operands, V_WMMA_F32_16X16X32_BF16, fp32 accumulate.
// Dominant GEMM (logits, 67 GFLOP) is HBM/L2-bound -> operands pre-converted
// to bf16 once; tiles staged to LDS with async DMA (ASYNCcnt) when available.
// ---------------------------------------------------------------------------

typedef __attribute__((ext_vector_type(16))) __bf16 v16bf;
typedef __attribute__((ext_vector_type(8)))  float  v8f;
typedef __attribute__((ext_vector_type(4)))  int    v4i;

#if __has_builtin(__builtin_amdgcn_global_load_async_to_lds_b128) && \
    __has_builtin(__builtin_amdgcn_s_wait_asynccnt)
#define HAVE_ASYNC_LDS 1
#else
#define HAVE_ASYNC_LDS 0
#endif

union Frag16 {            // 16 bf16 = 8 dwords = one WMMA A/B operand (wave32)
    v16bf    v;
    unsigned u[8];
    uint4    q[2];
};

__device__ __forceinline__ unsigned short f2bf(float f) {
    unsigned u = __float_as_uint(f);
    u += 0x7FFFu + ((u >> 16) & 1u);     // round-to-nearest-even
    return (unsigned short)(u >> 16);
}
__device__ __forceinline__ unsigned packbf(float lo, float hi) {
    return (unsigned)f2bf(lo) | ((unsigned)f2bf(hi) << 16);
}

#if HAVE_ASYNC_LDS
typedef __attribute__((address_space(1))) v4i gv4i;   // global int4
typedef __attribute__((address_space(3))) v4i lv4i;   // LDS int4
__device__ __forceinline__ void async_cp16(const unsigned short* g,
                                           unsigned short* l) {
    __builtin_amdgcn_global_load_async_to_lds_b128(
        (gv4i*)(const void*)g, (lv4i*)(void*)l, 0, 0);
}
#endif

// ---------------------------------------------------------------------------
// 1) Embedding gather straight to bf16 A-operand: emb_bf[b*T+t,:] = embed[x]
// ---------------------------------------------------------------------------
__global__ void gather_embed(const int* __restrict__ x,
                             const float* __restrict__ embed,
                             unsigned short* __restrict__ emb_bf,
                             int rows, int E) {
    int i = blockIdx.x * blockDim.x + threadIdx.x;
    if (i >= rows * E) return;
    int r = i / E, e = i - r * E;
    emb_bf[i] = f2bf(embed[(size_t)x[r] * E + e]);
}

// ---------------------------------------------------------------------------
// 2) fp32 -> bf16 bulk convert (weights), float4 -> uint2 vectorized.
// ---------------------------------------------------------------------------
__global__ void conv_bf16(const float* __restrict__ src,
                          unsigned short* __restrict__ dst, int n4) {
    int i = blockIdx.x * blockDim.x + threadIdx.x;
    if (i >= n4) return;
    float4 v = ((const float4*)src)[i];
    uint2 o;
    o.x = packbf(v.x, v.y);
    o.y = packbf(v.z, v.w);
    ((uint2*)dst)[i] = o;
}

// ---------------------------------------------------------------------------
// 3) W_hh -> bf16 WMMA B-fragment order for the RNN scan:
//    flat = ((ntile*16 + kchunk)*32 + lane)*16 + elem
//         = W_hh[kchunk*32 + (elem>>3)*16 + (lane>>4)*8 + (elem&7)][ntile*16 + (lane&15)]
// ---------------------------------------------------------------------------
__global__ void prep_whh_frag(const float* __restrict__ W,
                              unsigned short* __restrict__ F) {
    int i = blockIdx.x * blockDim.x + threadIdx.x;
    if (i >= 512 * 512) return;
    int e    = i & 15;
    int lane = (i >> 4) & 31;
    int frag = i >> 9;
    int kc   = frag & 15;
    int nt   = frag >> 4;
    F[i] = f2bf(W[(size_t)(kc * 32 + ((e >> 3) << 4) + ((lane >> 4) << 3) + (e & 7)) * 512 + nt * 16 + (lane & 15)]);
}

// ---------------------------------------------------------------------------
// 4) GEMM: C[M,N] = A[M,K](bf16) @ B[K,N](bf16) (+bias0+bias1), fp32 out.
//    M%128==0, N%128==0, K%32==0. 256 thr = 8 waves, wave grid 4(M) x 2(N),
//    2x4 WMMA tiles/wave. Double-buffered LDS; async DMA staging when the
//    gfx1250 GLOBAL_LOAD_ASYNC_TO_LDS builtins are available.
// ---------------------------------------------------------------------------
#define GBM 128
#define GBN 128
#define GBK 32
#define A_STR 40     // bf16/row: 80B rows keep 16B alignment; 20dw bank stride
#define B_STR 136    // 272B rows; 2-way conflict between lane L and L+16 only

__global__ __launch_bounds__(256) void gemm_bf16(
    const unsigned short* __restrict__ A, const unsigned short* __restrict__ B,
    const float* __restrict__ bias0, const float* __restrict__ bias1,
    float* __restrict__ C, int M, int N, int K)
{
    __shared__ alignas(16) unsigned short As[2][GBM * A_STR];
    __shared__ alignas(16) unsigned short Bs[2][GBK * B_STR];

    const int tid  = threadIdx.x;
    const int lane = tid & 31;
    const int w    = tid >> 5;
    const int wm   = w & 3;          // M offset wm*32
    const int wn   = w >> 2;         // N offset wn*64
    const int hf   = lane >> 4;
    const int l15  = lane & 15;
    const int m0   = blockIdx.y * GBM;
    const int n0   = blockIdx.x * GBN;
    const int KT   = K / GBK;

    // Per-thread staging slots: A tile 128x32 bf16 = 512 x 16B chunks,
    // B tile 32x128 bf16 = 512 x 16B chunks; 2 chunks each per thread.
    const int fA0 = tid,  fA1 = tid + 256;
    const int rA0 = fA0 >> 2,  cA0 = (fA0 & 3) << 3;
    const int rA1 = fA1 >> 2,  cA1 = (fA1 & 3) << 3;
    const int rB0 = fA0 >> 4,  cB0 = (fA0 & 15) << 3;
    const int rB1 = fA1 >> 4,  cB1 = (fA1 & 15) << 3;

    v8f acc[2][4];
#pragma unroll
    for (int mi = 0; mi < 2; ++mi)
#pragma unroll
        for (int ni = 0; ni < 4; ++ni)
#pragma unroll
            for (int r = 0; r < 8; ++r) acc[mi][ni][r] = 0.f;

#if HAVE_ASYNC_LDS
    // ---- async DMA double buffer ------------------------------------------
    {   // prologue: tile 0 -> buf 0
        async_cp16(A + (size_t)(m0 + rA0) * K + cA0, &As[0][rA0 * A_STR + cA0]);
        async_cp16(A + (size_t)(m0 + rA1) * K + cA1, &As[0][rA1 * A_STR + cA1]);
        async_cp16(B + (size_t)rB0 * N + n0 + cB0,   &Bs[0][rB0 * B_STR + cB0]);
        async_cp16(B + (size_t)rB1 * N + n0 + cB1,   &Bs[0][rB1 * B_STR + cB1]);
    }
    for (int kt = 0; kt < KT; ++kt) {
        const int cur = kt & 1;
        const int k0n = (kt + 1) * GBK;
        if (kt + 1 < KT) {
            const int nxt = cur ^ 1;
            async_cp16(A + (size_t)(m0 + rA0) * K + k0n + cA0, &As[nxt][rA0 * A_STR + cA0]);
            async_cp16(A + (size_t)(m0 + rA1) * K + k0n + cA1, &As[nxt][rA1 * A_STR + cA1]);
            async_cp16(B + (size_t)(k0n + rB0) * N + n0 + cB0, &Bs[nxt][rB0 * B_STR + cB0]);
            async_cp16(B + (size_t)(k0n + rB1) * N + n0 + cB1, &Bs[nxt][rB1 * B_STR + cB1]);
            __builtin_amdgcn_s_wait_asynccnt(4);   // current tile landed
        } else {
            __builtin_amdgcn_s_wait_asynccnt(0);
        }
        __syncthreads();
        // fragments + 8 WMMAs
        Frag16 af[2], bfg[4];
#pragma unroll
        for (int mi = 0; mi < 2; ++mi) {
            int row = wm * 32 + mi * 16 + l15;
#pragma unroll
            for (int i = 0; i < 8; ++i) {
                int kk = ((i >> 2) << 4) + (hf << 3) + ((i & 3) << 1);
                af[mi].u[i] = *(const unsigned*)&As[cur][row * A_STR + kk];
            }
        }
#pragma unroll
        for (int ni = 0; ni < 4; ++ni) {
            int nb = wn * 64 + ni * 16;
#pragma unroll
            for (int i = 0; i < 8; ++i) {
                int kk = ((i >> 2) << 4) + (hf << 3) + ((i & 3) << 1);
                bfg[ni].u[i] = (unsigned)Bs[cur][kk * B_STR + nb + l15] | ((unsigned)Bs[cur][(kk + 1) * B_STR + nb + l15] << 16);
            }
        }
#pragma unroll
        for (int mi = 0; mi < 2; ++mi)
#pragma unroll
            for (int ni = 0; ni < 4; ++ni)
                acc[mi][ni] = __builtin_amdgcn_wmma_f32_16x16x32_bf16(
                    false, af[mi].v, false, bfg[ni].v,
                    (short)0, acc[mi][ni], false, false);
        __syncthreads();   // reads done before next iteration's DMA into cur
    }
#else
    // ---- synchronous register double buffer -------------------------------
    {   // prologue: tile 0 -> buf 0
        uint4 a0 = *(const uint4*)(A + (size_t)(m0 + rA0) * K + cA0);
        uint4 a1 = *(const uint4*)(A + (size_t)(m0 + rA1) * K + cA1);
        uint4 b0 = *(const uint4*)(B + (size_t)rB0 * N + n0 + cB0);
        uint4 b1 = *(const uint4*)(B + (size_t)rB1 * N + n0 + cB1);
        *(uint4*)&As[0][rA0 * A_STR + cA0] = a0;
        *(uint4*)&As[0][rA1 * A_STR + cA1] = a1;
        *(uint4*)&Bs[0][rB0 * B_STR + cB0] = b0;
        *(uint4*)&Bs[0][rB1 * B_STR + cB1] = b1;
    }
    __syncthreads();
    for (int kt = 0; kt < KT; ++kt) {
        const int cur = kt & 1;
        const int k0n = (kt + 1) * GBK;
        uint4 a0, a1, b0, b1;
        if (kt + 1 < KT) {
            a0 = *(const uint4*)(A + (size_t)(m0 + rA0) * K + k0n + cA0);
            a1 = *(const uint4*)(A + (size_t)(m0 + rA1) * K + k0n + cA1);
            b0 = *(const uint4*)(B + (size_t)(k0n + rB0) * N + n0 + cB0);
            b1 = *(const uint4*)(B + (size_t)(k0n + rB1) * N + n0 + cB1);
        }
        Frag16 af[2], bfg[4];
#pragma unroll
        for (int mi = 0; mi < 2; ++mi) {
            int row = wm * 32 + mi * 16 + l15;
#pragma unroll
            for (int i = 0; i < 8; ++i) {
                int kk = ((i >> 2) << 4) + (hf << 3) + ((i & 3) << 1);
                af[mi].u[i] = *(const unsigned*)&As[cur][row * A_STR + kk];
            }
        }
#pragma unroll
        for (int ni = 0; ni < 4; ++ni) {
            int nb = wn * 64 + ni * 16;
#pragma unroll
            for (int i = 0; i < 8; ++i) {
                int kk = ((i >> 2) << 4) + (hf << 3) + ((i & 3) << 1);
                bfg[ni].u[i] = (unsigned)Bs[cur][kk * B_STR + nb + l15] | ((unsigned)Bs[cur][(kk + 1) * B_STR + nb + l15] << 16);
            }
        }
#pragma unroll
        for (int mi = 0; mi < 2; ++mi)
#pragma unroll
            for (int ni = 0; ni < 4; ++ni)
                acc[mi][ni] = __builtin_amdgcn_wmma_f32_16x16x32_bf16(
                    false, af[mi].v, false, bfg[ni].v,
                    (short)0, acc[mi][ni], false, false);
        if (kt + 1 < KT) {
            const int nxt = cur ^ 1;
            *(uint4*)&As[nxt][rA0 * A_STR + cA0] = a0;
            *(uint4*)&As[nxt][rA1 * A_STR + cA1] = a1;
            *(uint4*)&Bs[nxt][rB0 * B_STR + cB0] = b0;
            *(uint4*)&Bs[nxt][rB1 * B_STR + cB1] = b1;
        }
        __syncthreads();
    }
#endif

    // Epilogue: C layout vgpr r -> M = r + 8*half, N = lane&15.
#pragma unroll
    for (int mi = 0; mi < 2; ++mi)
#pragma unroll
        for (int ni = 0; ni < 4; ++ni) {
            int colg = n0 + wn * 64 + ni * 16 + l15;
            float badd = 0.f;
            if (bias0) badd += bias0[colg];
            if (bias1) badd += bias1[colg];
#pragma unroll
            for (int r = 0; r < 8; ++r) {
                int rowg = m0 + wm * 32 + mi * 16 + r + (hf << 3);
                C[(size_t)rowg * N + colg] = acc[mi][ni][r] + badd;
            }
        }
}

// ---------------------------------------------------------------------------
// 5) Sequential RNN scan (1 workgroup, 16 waves): h = tanh(h @ W_hh + e_all_t)
//    h kept in LDS as bf16 (A-fragment addressing); W_hh in fragment order.
// ---------------------------------------------------------------------------
#define H_STR 514

__global__ __launch_bounds__(512) void rnn_scan(
    const float* __restrict__ e_all,
    const unsigned short* __restrict__ WhhF,
    const float* __restrict__ h0,
    float* __restrict__ rnn_out,            // fp32 for attention
    unsigned short* __restrict__ rnn_bf,    // bf16 A-operand for proj GEMMs
    float* __restrict__ h_last)
{
    __shared__ unsigned short hbuf[16 * H_STR];
    const int tid  = threadIdx.x;
    const int lane = tid & 31;
    const int w    = tid >> 5;
    const int hf   = lane >> 4;
    const int l15  = lane & 15;

    for (int i = tid; i < 16 * H_STR; i += 512) hbuf[i] = 0;
    __syncthreads();
    for (int i = tid; i < 4 * 512; i += 512) {
        int b = i >> 9, n = i & 511;
        hbuf[b * H_STR + n] = f2bf(h0[i]);
    }
    __syncthreads();

    for (int t = 0; t < 256; ++t) {
        v8f acc[2];
#pragma unroll
        for (int j = 0; j < 2; ++j)
#pragma unroll
            for (int r = 0; r < 8; ++r) acc[j][r] = 0.f;

        for (int kc = 0; kc < 16; ++kc) {
            Frag16 af;
#pragma unroll
            for (int i = 0; i < 8; ++i) {
                int kk = kc * 32 + ((i >> 2) << 4) + (hf << 3) + ((i & 3) << 1);
                af.u[i] = *(const unsigned*)&hbuf[l15 * H_STR + kk];
            }
#pragma unroll
            for (int j = 0; j < 2; ++j) {
                int nt = w * 2 + j;
                Frag16 bfg;
                const uint4* p = (const uint4*)(WhhF +
                    ((size_t)((nt * 16 + kc) * 32 + lane) << 4));
                bfg.q[0] = p[0];
                bfg.q[1] = p[1];
                acc[j] = __builtin_amdgcn_wmma_f32_16x16x32_bf16(
                    false, af.v, false, bfg.v, (short)0, acc[j], false, false);
            }
        }
        __syncthreads();
#pragma unroll
        for (int j = 0; j < 2; ++j) {
            int n = (w * 2 + j) * 16 + l15;
#pragma unroll
            for (int r = 0; r < 8; ++r) {
                int Mrow = r + (hf << 3);
                if (Mrow < 4) {
                    size_t rowi = (size_t)(Mrow * 256 + t);
                    float vv = tanhf(acc[j][r] + e_all[rowi * 512 + n]);
                    unsigned short hb = f2bf(vv);
                    rnn_out[rowi * 512 + n] = vv;
                    rnn_bf [rowi * 512 + n] = hb;
                    hbuf[Mrow * H_STR + n]  = hb;
                    if (t == 255) h_last[Mrow * 512 + n] = vv;
                }
            }
        }
        __syncthreads();
    }
}

// ---------------------------------------------------------------------------
// 6) Fused Bahdanau row: scores -> causal softmax -> context; emits
//    combined = [rnn_out | context] directly in bf16 for the final GEMM.
// ---------------------------------------------------------------------------
__global__ __launch_bounds__(256) void attn_ctx(
    const float* __restrict__ proj_enc, const float* __restrict__ proj_dec,
    const float* __restrict__ attn_v,   const float* __restrict__ rnn_out,
    unsigned short* __restrict__ comb_bf)
{
    __shared__ float pd[512];
    __shared__ float vv[512];
    __shared__ float sc[256];
    __shared__ float red[16];

    const int tid = threadIdx.x, lane = tid & 31, w = tid >> 5;
    const int b = blockIdx.x >> 8, d = blockIdx.x & 255;
    const size_t rowd = (size_t)(b * 256 + d);

    pd[tid]       = proj_dec[rowd * 512 + tid];
    pd[tid + 256] = proj_dec[rowd * 512 + tid + 256];
    vv[tid]       = attn_v[tid];
    vv[tid + 256] = attn_v[tid + 256];
    __syncthreads();

    for (int e = w; e <= d; e += 8) {
        const float* pe = proj_enc + (size_t)(b * 256 + e) * 512;
        float p = 0.f;
        for (int h = lane; h < 512; h += 32)
            p += vv[h] * tanhf(pe[h] + pd[h]);
#pragma unroll
        for (int off = 16; off; off >>= 1) p += __shfl_xor(p, off, 32);
        if (lane == 0) sc[e] = p;
    }
    __syncthreads();

    float s = (tid <= d) ? sc[tid] : -1e30f;
    float m = s;
#pragma unroll
    for (int off = 16; off; off >>= 1) m = fmaxf(m, __shfl_xor(m, off, 32));
    if (lane == 0) red[w] = m;
    __syncthreads();
    if (tid == 0) {
        float mm = red[0];
        for (int i = 1; i < 8; ++i) mm = fmaxf(mm, red[i]);
        red[8] = mm;
    }
    __syncthreads();
    m = red[8];
    float p = (tid <= d) ? __expf(s - m) : 0.f;
    float ss = p;
#pragma unroll
    for (int off = 16; off; off >>= 1) ss += __shfl_xor(ss, off, 32);
    if (lane == 0) red[w] = ss;
    __syncthreads();
    if (tid == 0) {
        float tsum = 0.f;
        for (int i = 0; i < 8; ++i) tsum += red[i];
        red[9] = tsum;
    }
    __syncthreads();
    sc[tid] = p / red[9];
    __syncthreads();

    float c0 = 0.f, c1 = 0.f;
    for (int e = 0; e <= d; ++e) {
        float we = sc[e];
        const float* rr = rnn_out + (size_t)(b * 256 + e) * 512;
        c0 += we * rr[tid];
        c1 += we * rr[tid + 256];
    }
    unsigned short* crow = comb_bf + rowd * 1024;
    const float* rdr = rnn_out + rowd * 512;
    crow[tid]             = f2bf(rdr[tid]);
    crow[tid + 256]       = f2bf(rdr[tid + 256]);
    crow[512 + tid]       = f2bf(c0);
    crow[512 + 256 + tid] = f2bf(c1);
}

// ---------------------------------------------------------------------------
// Launch: inputs in setup_inputs() order.
// ---------------------------------------------------------------------------
extern "C" void kernel_launch(void* const* d_in, const int* in_sizes, int n_in,
                              void* d_out, int out_size, void* d_ws, size_t ws_size,
                              hipStream_t stream) {
    (void)in_sizes; (void)n_in; (void)out_size; (void)ws_size;
    const int*   x      = (const int*)  d_in[0];
    const float* embed  = (const float*)d_in[1];
    const float* W_ih   = (const float*)d_in[2];
    const float* W_hh   = (const float*)d_in[3];
    const float* b_ih   = (const float*)d_in[4];
    const float* b_hh   = (const float*)d_in[5];
    const float* attn_W = (const float*)d_in[6];
    const float* attn_U = (const float*)d_in[7];
    const float* attn_v = (const float*)d_in[8];
    const float* fc_W   = (const float*)d_in[9];
    const float* fc_b   = (const float*)d_in[10];
    const float* h0     = (const float*)d_in[11];
    float* out = (float*)d_out;

    char* ws = (char*)d_ws;
    const size_t MB = 1u << 20;
    float*          e_all   = (float*)(ws + 0 * MB);            // 2 MB
    float*          rnn     = (float*)(ws + 2 * MB);            // 2 MB
    float*          penc    = (float*)(ws + 4 * MB);            // 2 MB
    float*          pdec    = (float*)(ws + 6 * MB);            // 2 MB
    unsigned short* emb_bf  = (unsigned short*)(ws + 8 * MB);   // 1 MB
    unsigned short* rnn_bf  = (unsigned short*)(ws + 9 * MB);   // 1 MB
    unsigned short* comb_bf = (unsigned short*)(ws + 10 * MB);  // 2 MB
    unsigned short* whhF    = (unsigned short*)(ws + 12 * MB);  // 0.5 MB
    unsigned short* wih_bf  = (unsigned short*)(ws + (12 * MB + 512 * 1024));
    unsigned short* attW_bf = (unsigned short*)(ws + 13 * MB);
    unsigned short* attU_bf = (unsigned short*)(ws + (13 * MB + 512 * 1024));
    unsigned short* fcW_bf  = (unsigned short*)(ws + 14 * MB);  // 64 MB

    const int BT = 1024;   // B*T
    // 1) gather (bf16)
    gather_embed<<<(BT * 512 + 255) / 256, 256, 0, stream>>>(x, embed, emb_bf, BT, 512);
    // 2) weight conversions
    prep_whh_frag<<<(512 * 512) / 256, 256, 0, stream>>>(W_hh, whhF);
    conv_bf16<<<(512 * 512 / 4) / 256, 256, 0, stream>>>(W_ih,   wih_bf,  512 * 512 / 4);
    conv_bf16<<<(512 * 512 / 4) / 256, 256, 0, stream>>>(attn_W, attW_bf, 512 * 512 / 4);
    conv_bf16<<<(512 * 512 / 4) / 256, 256, 0, stream>>>(attn_U, attU_bf, 512 * 512 / 4);
    conv_bf16<<<(1024 * 32000 / 4) / 256, 256, 0, stream>>>(fc_W, fcW_bf, 1024 * 32000 / 4);
    // 3) hoisted input GEMM: e_all = emb @ W_ih + (b_ih + b_hh)
    gemm_bf16<<<dim3(512 / GBN, BT / GBM), 256, 0, stream>>>(
        emb_bf, wih_bf, b_ih, b_hh, e_all, BT, 512, 512);
    // 4) sequential scan (also writes h_last at tail of d_out)
    rnn_scan<<<1, 512, 0, stream>>>(e_all, whhF, h0, rnn, rnn_bf,
                                    out + (size_t)4 * 256 * 32000);
    // 5) attention projections
    gemm_bf16<<<dim3(512 / GBN, BT / GBM), 256, 0, stream>>>(
        rnn_bf, attW_bf, nullptr, nullptr, penc, BT, 512, 512);
    gemm_bf16<<<dim3(512 / GBN, BT / GBM), 256, 0, stream>>>(
        rnn_bf, attU_bf, nullptr, nullptr, pdec, BT, 512, 512);
    // 6) fused scores/softmax/context -> combined (bf16)
    attn_ctx<<<BT, 256, 0, stream>>>(penc, pdec, attn_v, rnn, comb_bf);
    // 7) dominant GEMM: logits = combined @ fc_W + fc_b
    gemm_bf16<<<dim3(32000 / GBN, BT / GBM), 256, 0, stream>>>(
        comb_bf, fcW_bf, fc_b, nullptr, out, BT, 32000, 1024);
}
